// MVRNNAnomalyQuality_10385230922081
// MI455X (gfx1250) — compile-verified, matrix operand order checked by reading, not run on hardware
//
#include <hip/hip_runtime.h>

// ---------------------------------------------------------------------------
// VRNN (GRU+LSTM hybrid) forward losses on gfx1250.
//   B=128, S=1024, D=L=256.
//  Phase 1 (whole-chip throughput): x_enc[t] = relu(x_t @ enc_phi_W.T + b)
//     for all t, as a big WMMA GEMM into a time-major f16 buffer (if ws fits).
//  Phase 2 (latency-bound recurrence): 16 persistent workgroups, each owning
//     16 latent/output columns.
//   - Per-WG weight tiles (31 x [256K x 16N] f16 = 248KB) async-copied into
//     LDS once (GLOBAL_LOAD_ASYNC_TO_LDS_B128) and resident for all steps.
//   - Every GEMM is v_wmma_f32_16x16x32_f16; M=128 = 8 waves x 16 rows.
//   - h/z/z_enc slices exchanged via f16 scratch in L2; atomic-counter grid
//     barrier (agent scope) 3x per step; prefetch of next step's x/eps.
//   - h,c accumulator tiles stay in registers in WMMA C layout across steps.
// ---------------------------------------------------------------------------

typedef __attribute__((ext_vector_type(16))) _Float16 hv16;
typedef __attribute__((ext_vector_type(8)))  _Float16 hv8;
typedef __attribute__((ext_vector_type(8)))  float    fv8;
typedef int vint4 __attribute__((vector_size(16)));

#define S_LEN 1024
#define BATCH 128
#define LAT   256
#define DIN   256
#define NWG   16
#define NSLOT 31
#define TILE_HALFS 4096              /* 8 kfrags * 32 lanes * 16 halves */
#define SMEM_HALFS (NSLOT * TILE_HALFS)

// slot map (per WG, c0 = 16*wg):
//  0        : enc_phi_W             (x_t -> x_enc)                 stage 0 / phase 1
//  1..2     : prior_W    mu/ls      (h)                            stage A
//  3..4     : enc_W[:, :L]  mu/ls   (x_enc)                        stage A
//  5..6     : enc_W[:, L:]  mu/ls   (h)                            stage A
//  7..9     : gru_Whh   r/z/n       (h)                            stage A
// 10..12    : gru_Wih[:, :L] r/z/n  (x_enc)                        stage A
// 13..16    : lstm_Whh  i/f/g/o     (h)                            stage A
// 17..20    : lstm_Wih[:, :L]       (x_enc)                        stage A
// 21        : dec_W[:, L:]          (h)                            stage A
// 22        : dec_phi_W             (z -> z_enc)                   stage B
// 23        : dec_W[:, :L]          (z_enc)                        stage C
// 24..26    : gru_Wih[:, L:] r/z/n  (z_enc)                        stage C
// 27..30    : lstm_Wih[:, L:]       (z_enc)                        stage C

constexpr int SLH[12] = {1, 2, 5, 6, 7, 8, 9, 13, 14, 15, 16, 21};
constexpr int SLX[9]  = {3, 4, 10, 11, 12, 17, 18, 19, 20};
constexpr int SLZ[8]  = {23, 24, 25, 26, 27, 28, 29, 30};

__device__ __forceinline__ fv8 vzero8() {
  fv8 v = {0.f, 0.f, 0.f, 0.f, 0.f, 0.f, 0.f, 0.f};
  return v;
}

__device__ __forceinline__ fv8 wmma16(hv16 a, hv16 b, fv8 c) {
  return __builtin_amdgcn_wmma_f32_16x16x32_f16(
      /*neg_a=*/false, a, /*neg_b=*/false, b,
      /*c_mod=*/(short)0, c, /*reuse_a=*/false, /*reuse_b=*/false);
}

// A-fragment (16x32 f16) from row-major f16 array (ld = LAT), per ISA 7.12.2:
//  lanes 0-15: row M=lane,    K in [k0,k0+8) and [k0+16,k0+24)
//  lanes16-31: row M=lane-16, K in [k0+8,k0+16) and [k0+24,k0+32)
__device__ __forceinline__ hv16 loadA_f16(const _Float16* __restrict__ base,
                                          int row0, int k0, int lane) {
  const int lr = lane & 15, hi = lane >> 4;
  const _Float16* p = base + (size_t)(row0 + lr) * LAT + k0 + hi * 8;
  hv16 a;
  *(hv8*)&a       = *(const hv8*)p;
  *((hv8*)&a + 1) = *(const hv8*)(p + 16);
  return a;
}

// Same A layout but sourced from an f32 array with arbitrary row stride.
__device__ __forceinline__ hv16 loadA_f32(const float* __restrict__ base,
                                          size_t ld, int row0, int k0, int lane) {
  const int lr = lane & 15, hi = lane >> 4;
  const float* p = base + (size_t)(row0 + lr) * ld + k0 + hi * 8;
  hv16 a;
  _Float16* ah = (_Float16*)&a;
#pragma unroll
  for (int i = 0; i < 8; ++i) ah[i] = (_Float16)p[i];
#pragma unroll
  for (int i = 0; i < 8; ++i) ah[8 + i] = (_Float16)p[16 + i];
  return a;
}

// B-fragment (pre-swizzled: 16 contiguous halves per lane per kfrag).
__device__ __forceinline__ hv16 loadB(const _Float16* __restrict__ wbase,
                                      int slot, int kf, int lane) {
  const _Float16* p = wbase + ((size_t)((slot * 8 + kf) * 32 + lane) * 16);
  hv16 b;
  *(hv8*)&b       = *(const hv8*)p;
  *((hv8*)&b + 1) = *(const hv8*)(p + 8);
  return b;
}

// K=256 GEMM over a group of output tiles sharing the same A operand.
template <int N>
__device__ __forceinline__ void gemm_group(fv8* acc, const int (&slots)[N],
                                           const _Float16* __restrict__ smemW,
                                           const _Float16* __restrict__ Abase,
                                           int row0, int lane) {
#pragma unroll
  for (int kf = 0; kf < 8; ++kf) {
    hv16 a = loadA_f16(Abase, row0, kf * 32, lane);
#pragma unroll
    for (int i = 0; i < N; ++i) {
      hv16 b = loadB(smemW, slots[i], kf, lane);
      acc[i] = wmma16(a, b, acc[i]);
    }
  }
}

__device__ __forceinline__ float sigmoidf(float v) {
  return 1.f / (1.f + __expf(-v));
}

__device__ __forceinline__ void grid_sync(unsigned* bar, int* epoch) {
  __syncthreads();
  if (threadIdx.x == 0) {
    ++(*epoch);
    __threadfence();
    __hip_atomic_fetch_add(bar, 1u, __ATOMIC_ACQ_REL, __HIP_MEMORY_SCOPE_AGENT);
    const unsigned target = (unsigned)(*epoch) * NWG;
    while (__hip_atomic_load(bar, __ATOMIC_ACQUIRE, __HIP_MEMORY_SCOPE_AGENT) < target) {
      __builtin_amdgcn_s_sleep(1);
    }
  }
  __syncthreads();
}

// --------------------------- prep kernels ----------------------------------

__global__ void vrnn_zero(unsigned* bar, float* accums) {
  if (threadIdx.x == 0) {
    *bar = 0u;
    accums[0] = 0.f;   // kl sum
    accums[1] = 0.f;   // bce sum
  }
}

__device__ __forceinline__ void slot_desc(
    int slot, int c0,
    const float* prior_W, const float* enc_phi_W, const float* enc_W,
    const float* dec_phi_W, const float* dec_W,
    const float* gru_Wih, const float* gru_Whh,
    const float* lstm_Wih, const float* lstm_Whh,
    const float** W, int* ldw, int* n0, int* kofs) {
  if (slot == 0)       { *W = enc_phi_W; *ldw = 256; *n0 = c0;                   *kofs = 0;   }
  else if (slot <= 2)  { *W = prior_W;   *ldw = 256; *n0 = c0 + 256*(slot - 1);  *kofs = 0;   }
  else if (slot <= 4)  { *W = enc_W;     *ldw = 512; *n0 = c0 + 256*(slot - 3);  *kofs = 0;   }
  else if (slot <= 6)  { *W = enc_W;     *ldw = 512; *n0 = c0 + 256*(slot - 5);  *kofs = 256; }
  else if (slot <= 9)  { *W = gru_Whh;   *ldw = 256; *n0 = c0 + 256*(slot - 7);  *kofs = 0;   }
  else if (slot <= 12) { *W = gru_Wih;   *ldw = 512; *n0 = c0 + 256*(slot - 10); *kofs = 0;   }
  else if (slot <= 16) { *W = lstm_Whh;  *ldw = 256; *n0 = c0 + 256*(slot - 13); *kofs = 0;   }
  else if (slot <= 20) { *W = lstm_Wih;  *ldw = 512; *n0 = c0 + 256*(slot - 17); *kofs = 0;   }
  else if (slot == 21) { *W = dec_W;     *ldw = 512; *n0 = c0;                   *kofs = 256; }
  else if (slot == 22) { *W = dec_phi_W; *ldw = 256; *n0 = c0;                   *kofs = 0;   }
  else if (slot == 23) { *W = dec_W;     *ldw = 512; *n0 = c0;                   *kofs = 0;   }
  else if (slot <= 26) { *W = gru_Wih;   *ldw = 512; *n0 = c0 + 256*(slot - 24); *kofs = 256; }
  else                 { *W = lstm_Wih;  *ldw = 512; *n0 = c0 + 256*(slot - 27); *kofs = 256; }
}

// Convert weights f32 -> f16 and swizzle into WMMA B-fragment lane layout.
__global__ void vrnn_pack(
    const float* __restrict__ prior_W, const float* __restrict__ enc_phi_W,
    const float* __restrict__ enc_W, const float* __restrict__ dec_phi_W,
    const float* __restrict__ dec_W, const float* __restrict__ gru_Wih,
    const float* __restrict__ gru_Whh, const float* __restrict__ lstm_Wih,
    const float* __restrict__ lstm_Whh, _Float16* __restrict__ wpack) {
  const int wg = blockIdx.x / NSLOT;
  const int slot = blockIdx.x % NSLOT;
  const int c0 = wg * 16;
  const float* W; int ldw, n0, kofs;
  slot_desc(slot, c0, prior_W, enc_phi_W, enc_W, dec_phi_W, dec_W,
            gru_Wih, gru_Whh, lstm_Wih, lstm_Whh, &W, &ldw, &n0, &kofs);
  _Float16* dst = wpack + ((size_t)wg * NSLOT + slot) * TILE_HALFS;
  for (int e = threadIdx.x; e < TILE_HALFS; e += blockDim.x) {
    const int kf = e >> 9;          // k-fragment 0..7
    const int rem = e & 511;
    const int ln = rem >> 4;        // lane 0..31
    const int j2 = rem & 15;        // half index within lane
    const int j = j2 >> 1, ip = j2 & 1;
    const int kk = ((j < 4) ? 2 * j : 16 + 2 * (j - 4)) + ip + ((ln >= 16) ? 8 : 0);
    const int nn = ln & 15;         // column within tile
    const int K = kofs + kf * 32 + kk;
    dst[e] = (_Float16)W[(size_t)(n0 + nn) * ldw + K];
  }
}

// ---- Phase 1: x_enc for ALL timesteps (throughput GEMM, whole chip) -------
__global__ void __launch_bounds__(256, 1) vrnn_phase1(
    const float* __restrict__ x, const float* __restrict__ enc_phi_b,
    const _Float16* __restrict__ wpack, _Float16* __restrict__ xenc_big) {
  const int t = blockIdx.x;
  const int tid = threadIdx.x, lane = tid & 31, wv = tid >> 5;
  const int row0 = wv * 16, hi = lane >> 4;
  // Convert this wave's 16 batch rows of x_t into A fragments once.
  hv16 afrag[8];
  const float* xB = x + (size_t)t * DIN;   // row stride S*D
#pragma unroll
  for (int kf = 0; kf < 8; ++kf)
    afrag[kf] = loadA_f32(xB, (size_t)S_LEN * DIN, row0, kf * 32, lane);
  _Float16* outB = xenc_big + (size_t)t * BATCH * LAT;
  for (int cb = 0; cb < NWG; ++cb) {        // 16 column tiles (L2-resident B)
    const _Float16* btile = wpack + ((size_t)cb * NSLOT) * TILE_HALFS; // slot 0
    fv8 acc = vzero8();
#pragma unroll
    for (int kf = 0; kf < 8; ++kf) {
      hv16 b = loadB(btile, 0, kf, lane);
      acc = wmma16(afrag[kf], b, acc);
    }
    const int col = cb * 16 + (lane & 15);
    const float bb = enc_phi_b[col];
#pragma unroll
    for (int r = 0; r < 8; ++r) {
      const int m = row0 + r + 8 * hi;
      outB[m * LAT + col] = (_Float16)fmaxf(acc[r] + bb, 0.f);
    }
  }
}

// --------------------------- main recurrent kernel -------------------------

__global__ void __launch_bounds__(256, 1) vrnn_phase2(
    const float* __restrict__ x, const float* __restrict__ eps,
    const float* __restrict__ prior_b, const float* __restrict__ enc_phi_b,
    const float* __restrict__ enc_b, const float* __restrict__ dec_phi_b,
    const float* __restrict__ dec_b, const float* __restrict__ gru_bih,
    const float* __restrict__ gru_bhh, const float* __restrict__ lstm_bih,
    const float* __restrict__ lstm_bhh, const float* __restrict__ alphaPtr,
    const _Float16* __restrict__ wpack,
    _Float16* __restrict__ xenc_scr, _Float16* __restrict__ z_scr,
    _Float16* __restrict__ zenc_scr, _Float16* __restrict__ h_scr,
    const _Float16* __restrict__ xenc_big, int use_pre,
    unsigned* bar, float* accums) {
  extern __shared__ _Float16 smemW[];     // 31 tiles * 8KB = 248KB
  __shared__ float redK[256];
  __shared__ float redR[256];

  const int tid  = threadIdx.x;
  const int wg   = blockIdx.x;
  const int lane = tid & 31;
  const int wv   = tid >> 5;
  const int row0 = wv * 16;               // batch rows [row0, row0+16)
  const int c0   = wg * 16;               // owned columns
  const int col  = c0 + (lane & 15);
  const int hi   = lane >> 4;

  // ---- Stage weights into LDS once (async LDS DMA when available) ----
#if __has_builtin(__builtin_amdgcn_global_load_async_to_lds_b128)
  {
    char* gsrc = (char*)(wpack + (size_t)wg * SMEM_HALFS);   // const cast away
    char* ldst = (char*)smemW;
    for (unsigned off = (unsigned)tid * 16u; off < SMEM_HALFS * 2u; off += 256u * 16u) {
      __builtin_amdgcn_global_load_async_to_lds_b128(
          (__attribute__((address_space(1))) vint4*)(gsrc + off),
          (__attribute__((address_space(3))) vint4*)(ldst + off),
          0, 0);
    }
#if __has_builtin(__builtin_amdgcn_s_wait_asynccnt)
    __builtin_amdgcn_s_wait_asynccnt(0);
#else
    asm volatile("s_wait_asynccnt 0" ::: "memory");
#endif
  }
#else
  {
    const uint4* s = (const uint4*)(wpack + (size_t)wg * SMEM_HALFS);
    uint4* d = (uint4*)smemW;
    for (int i = tid; i < SMEM_HALFS / 8; i += blockDim.x) d[i] = s[i];
  }
#endif
  // Zero own h slice (h0 = 0).
  for (int i = tid; i < BATCH * 16; i += blockDim.x) {
    const int m = i >> 4, cc = c0 + (i & 15);
    h_scr[m * LAT + cc] = (_Float16)0.f;
  }

  const float alpha = alphaPtr[0];

  // Per-thread bias scalars for owned column.
  const float b_pr_mu = prior_b[col],      b_pr_ls = prior_b[LAT + col];
  const float b_en_mu = enc_b[col],        b_en_ls = enc_b[LAT + col];
  const float b_phi   = enc_phi_b[col],    b_dphi  = dec_phi_b[col];
  const float b_dec   = dec_b[col];
  const float b_gih_r = gru_bih[col],      b_gih_z = gru_bih[LAT + col],
              b_gih_n = gru_bih[2 * LAT + col];
  const float b_ghh_r = gru_bhh[col],      b_ghh_z = gru_bhh[LAT + col],
              b_ghh_n = gru_bhh[2 * LAT + col];
  const float b_l_i = lstm_bih[col]           + lstm_bhh[col];
  const float b_l_f = lstm_bih[LAT + col]     + lstm_bhh[LAT + col];
  const float b_l_g = lstm_bih[2 * LAT + col] + lstm_bhh[2 * LAT + col];
  const float b_l_o = lstm_bih[3 * LAT + col] + lstm_bhh[3 * LAT + col];

  fv8 h_reg = vzero8();   // own h tile, WMMA C layout, persists across steps
  fv8 c_reg = vzero8();   // own c tile, never leaves registers
  float kl_acc = 0.f, rec_acc = 0.f;
  int epoch = 0;

  for (int t = 0; t < S_LEN; ++t) {
    // ---- Stage 0 (fallback only): x_enc[:, own cols] ----
    if (!use_pre) {
      fv8 a0 = vzero8();
      const float* xB = x + (size_t)t * DIN;   // row stride S*D
#pragma unroll
      for (int kf = 0; kf < 8; ++kf) {
        hv16 a = loadA_f32(xB, (size_t)S_LEN * DIN, row0, kf * 32, lane);
        hv16 b = loadB(smemW, 0, kf, lane);
        a0 = wmma16(a, b, a0);
      }
#pragma unroll
      for (int r = 0; r < 8; ++r) {
        const int m = row0 + r + 8 * hi;
        xenc_scr[m * LAT + col] = (_Float16)fmaxf(a0[r] + b_phi, 0.f);
      }
    }
    grid_sync(bar, &epoch);   // B0: x_enc + h(prev step) visible

    const _Float16* xsrc =
        use_pre ? (xenc_big + (size_t)t * BATCH * LAT) : xenc_scr;

    // ---- Stage A: everything depending on h and x_enc ----
    fv8 AH[12];
#pragma unroll
    for (int i = 0; i < 12; ++i) AH[i] = vzero8();
    gemm_group(AH, SLH, smemW, h_scr, row0, lane);
    fv8 AX[9];
#pragma unroll
    for (int i = 0; i < 9; ++i) AX[i] = vzero8();
    gemm_group(AX, SLX, smemW, xsrc, row0, lane);

    fv8 ghr = vzero8(), ghz = vzero8(), ghn = vzero8();
    fv8 gir = vzero8(), giz = vzero8(), gin = vzero8();
    fv8 g0 = vzero8(), g1 = vzero8(), g2 = vzero8(), g3 = vzero8();
    fv8 lgt = vzero8();
#pragma unroll
    for (int r = 0; r < 8; ++r) {
      const int m = row0 + r + 8 * hi;
      const float mu_pr = AH[0][r] + b_pr_mu;
      const float ls_pr = AH[1][r] + b_pr_ls;
      const float mu_po = AH[2][r] + AX[0][r] + b_en_mu;
      const float ls_po = AH[3][r] + AX[1][r] + b_en_ls;
      const float e = eps[((size_t)m * S_LEN + t) * LAT + col];
      const float zv = mu_po + __expf(ls_po) * e;
      z_scr[m * LAT + col] = (_Float16)zv;
      const float dmu = mu_po - mu_pr;
      const float t1 = 2.f * (ls_pr - ls_po);
      const float t2 = (__expf(2.f * ls_po) + dmu * dmu) * __expf(-2.f * ls_pr);
      kl_acc += 0.5f * (t1 + t2);
      ghr[r] = AH[4][r] + b_ghh_r;
      ghz[r] = AH[5][r] + b_ghh_z;
      ghn[r] = AH[6][r] + b_ghh_n;
      g0[r]  = AH[7][r]  + AX[5][r] + b_l_i;
      g1[r]  = AH[8][r]  + AX[6][r] + b_l_f;
      g2[r]  = AH[9][r]  + AX[7][r] + b_l_g;
      g3[r]  = AH[10][r] + AX[8][r] + b_l_o;
      lgt[r] = AH[11][r] + b_dec;
      gir[r] = AX[2][r] + b_gih_r;
      giz[r] = AX[3][r] + b_gih_z;
      gin[r] = AX[4][r] + b_gih_n;
    }
    grid_sync(bar, &epoch);   // B1: z visible

    // ---- Stage B: z_enc = relu(z @ dec_phi_W.T + b) ----
    {
      fv8 ze = vzero8();
#pragma unroll
      for (int kf = 0; kf < 8; ++kf) {
        hv16 a = loadA_f16(z_scr, row0, kf * 32, lane);
        hv16 b = loadB(smemW, 22, kf, lane);
        ze = wmma16(a, b, ze);
      }
#pragma unroll
      for (int r = 0; r < 8; ++r) {
        const int m = row0 + r + 8 * hi;
        zenc_scr[m * LAT + col] = (_Float16)fmaxf(ze[r] + b_dphi, 0.f);
      }
    }
    grid_sync(bar, &epoch);   // B2: z_enc visible

    // Prefetch next step's x/eps lines while GEMMs below run.
    if (t + 1 < S_LEN) {
      const int mp = row0 + (lane & 15);
      __builtin_prefetch(&x[((size_t)mp * S_LEN + t + 1) * DIN + col], 0, 1);
      __builtin_prefetch(&eps[((size_t)mp * S_LEN + t + 1) * LAT + col], 0, 1);
    }

    // ---- Stage C: z_enc-dependent GEMMs + gates + losses + h update ----
    fv8 AZ[8];
#pragma unroll
    for (int i = 0; i < 8; ++i) AZ[i] = vzero8();
    gemm_group(AZ, SLZ, smemW, zenc_scr, row0, lane);

#pragma unroll
    for (int r = 0; r < 8; ++r) {
      const int m = row0 + r + 8 * hi;
      // BCE-with-logits, sum reduction
      const float l = lgt[r] + AZ[0][r];
      const float xv = x[((size_t)m * S_LEN + t) * DIN + col];
      rec_acc += fmaxf(l, 0.f) - l * xv + log1pf(__expf(-fabsf(l)));
      // GRU branch
      const float rg = sigmoidf(gir[r] + AZ[1][r] + ghr[r]);
      const float zg = sigmoidf(giz[r] + AZ[2][r] + ghz[r]);
      const float ng = tanhf(gin[r] + AZ[3][r] + rg * ghn[r]);
      const float h_gru = (1.f - zg) * ng + zg * h_reg[r];
      // LSTM branch
      const float gi_ = g0[r] + AZ[4][r];
      const float gf_ = g1[r] + AZ[5][r];
      const float gg_ = g2[r] + AZ[6][r];
      const float go_ = g3[r] + AZ[7][r];
      const float c_new = sigmoidf(gf_) * c_reg[r] + sigmoidf(gi_) * tanhf(gg_);
      const float h_lstm = sigmoidf(go_) * tanhf(c_new);
      const float h_new = alpha * h_gru + alpha * h_lstm;
      h_reg[r] = h_new;
      c_reg[r] = c_new;
      h_scr[m * LAT + col] = (_Float16)h_new;
    }
    // next iteration's B0 publishes h_new
  }

  // ---- final reduction ----
  __syncthreads();
  redK[tid] = kl_acc;
  redR[tid] = rec_acc;
  __syncthreads();
  for (int off = 128; off > 0; off >>= 1) {
    if (tid < off) {
      redK[tid] += redK[tid + off];
      redR[tid] += redR[tid + off];
    }
    __syncthreads();
  }
  if (tid == 0) {
    atomicAdd(&accums[0], redK[0]);
    atomicAdd(&accums[1], redR[0]);
  }
}

__global__ void vrnn_finalize(const float* accums, float* out) {
  // KL analytic constant: sum over S steps, B batch of (-0.5 * L)
  const float kl_const = -0.5f * (float)LAT * (float)BATCH * (float)S_LEN;
  const float kl = (accums[0] + kl_const) / (float)(S_LEN * 10);
  const float rec = accums[1] / (float)(S_LEN * 10);
  out[0] = rec + kl;   // total
  out[1] = rec;        // recon_loss
  out[2] = kl;         // kl_div
}

// --------------------------- host launch -----------------------------------

extern "C" void kernel_launch(void* const* d_in, const int* in_sizes, int n_in,
                              void* d_out, int out_size, void* d_ws, size_t ws_size,
                              hipStream_t stream) {
  const float* x         = (const float*)d_in[0];
  const float* eps       = (const float*)d_in[1];
  const float* prior_W   = (const float*)d_in[2];
  const float* prior_b   = (const float*)d_in[3];
  const float* enc_phi_W = (const float*)d_in[4];
  const float* enc_phi_b = (const float*)d_in[5];
  const float* enc_W     = (const float*)d_in[6];
  const float* enc_b     = (const float*)d_in[7];
  const float* dec_phi_W = (const float*)d_in[8];
  const float* dec_phi_b = (const float*)d_in[9];
  const float* dec_W     = (const float*)d_in[10];
  const float* dec_b     = (const float*)d_in[11];
  const float* gru_Wih   = (const float*)d_in[12];
  const float* gru_Whh   = (const float*)d_in[13];
  const float* gru_bih   = (const float*)d_in[14];
  const float* gru_bhh   = (const float*)d_in[15];
  const float* lstm_Wih  = (const float*)d_in[16];
  const float* lstm_Whh  = (const float*)d_in[17];
  const float* lstm_bih  = (const float*)d_in[18];
  const float* lstm_bhh  = (const float*)d_in[19];
  const float* alpha     = (const float*)d_in[20];

  char* ws = (char*)d_ws;
  unsigned* bar  = (unsigned*)(ws + 0);
  float* accums  = (float*)(ws + 256);
  _Float16* wpack = (_Float16*)(ws + 512);
  const size_t wbytes = (size_t)NWG * NSLOT * TILE_HALFS * sizeof(_Float16);
  _Float16* xenc = (_Float16*)(ws + 512 + wbytes);
  _Float16* zsc  = xenc + (size_t)BATCH * LAT;
  _Float16* zenc = zsc  + (size_t)BATCH * LAT;
  _Float16* hsc  = zenc + (size_t)BATCH * LAT;
  const size_t small_bytes = 512 + wbytes + (size_t)4 * BATCH * LAT * sizeof(_Float16);
  const size_t big_bytes = (size_t)S_LEN * BATCH * LAT * sizeof(_Float16);
  const int use_pre = (ws_size >= small_bytes + big_bytes) ? 1 : 0;
  _Float16* xenc_big = (_Float16*)(ws + small_bytes);

  vrnn_zero<<<1, 64, 0, stream>>>(bar, accums);
  vrnn_pack<<<NWG * NSLOT, 256, 0, stream>>>(prior_W, enc_phi_W, enc_W,
                                             dec_phi_W, dec_W, gru_Wih, gru_Whh,
                                             lstm_Wih, lstm_Whh, wpack);
  if (use_pre) {
    vrnn_phase1<<<S_LEN, 256, 0, stream>>>(x, enc_phi_b, wpack, xenc_big);
  }
  vrnn_phase2<<<NWG, 256, SMEM_HALFS * sizeof(_Float16), stream>>>(
      x, eps, prior_b, enc_phi_b, enc_b, dec_phi_b, dec_b,
      gru_bih, gru_bhh, lstm_bih, lstm_bhh, alpha,
      wpack, xenc, zsc, zenc, hsc, xenc_big, use_pre, bar, accums);
  vrnn_finalize<<<1, 1, 0, stream>>>(accums, (float*)d_out);
}
